// SpatialViewAttention_30915174596611
// MI455X (gfx1250) — compile-verified, matrix operand order checked by reading, not run on hardware
//
#include <hip/hip_runtime.h>
#include <math.h>

// ---------------------------------------------------------------------------
// SpatialViewAttention for MI455X (gfx1250), wave32 + WMMA + TDM.
//
// Shapes: B=2, N=16384, V=8, E=256, H=W=64, NH=8, HD=32, M=B*N=32768.
// Exact algebraic simplifications used:
//   * q_in identical across V  -> attention context identical across V
//   * view_attention == 1/V exactly -> aggregated == final row per point
// All E x E projections run as v_wmma_f32_16x16x32_f16 (f16 in, f32 acc).
// A-tiles are staged into LDS with the Tensor Data Mover when available.
// ---------------------------------------------------------------------------

typedef _Float16 h8_t  __attribute__((ext_vector_type(8)));
typedef _Float16 v16h  __attribute__((ext_vector_type(16)));
typedef float    v8f   __attribute__((ext_vector_type(8)));

static constexpr int  CB  = 2;
static constexpr int  CN  = 16384;
static constexpr int  CV  = 8;
static constexpr int  CE  = 256;
static constexpr int  CH  = 64;
static constexpr int  CW  = 64;
static constexpr long CM  = (long)CB * CN;          // 32768 points
static constexpr long CMV = CM * CV;                // 262144 kv rows

// ---------------------------------------------------------------------------
// Tensor Data Mover: 1-D contiguous f16 tile load, global -> LDS.
// D# packing per CDNA5 ISA.
// ---------------------------------------------------------------------------
#if defined(__has_builtin)
#if __has_builtin(__builtin_amdgcn_tensor_load_to_lds) && \
    __has_builtin(__builtin_amdgcn_s_wait_tensorcnt)
#define HAVE_TDM 1
#endif
#endif

#ifdef HAVE_TDM
typedef unsigned int tdm_v4u __attribute__((ext_vector_type(4)));
typedef int          tdm_v4i __attribute__((ext_vector_type(4)));
typedef int          tdm_v8i __attribute__((ext_vector_type(8)));

// Generic LDS pointer -> 32-bit LDS byte offset (flat LDS aperture keeps the
// offset in ADDR[31:0], ISA 10.2).
__device__ __forceinline__ unsigned lds_off_of(const void* p) {
  return (unsigned)(unsigned long long)p;
}

// nelem f16 elements (contiguous, nelem <= 65535, multiple of 2) -> LDS.
__device__ __forceinline__ void tdm_load_1d_f16(unsigned lds_off,
                                                const void* gptr,
                                                unsigned nelem) {
  const unsigned long long ga = (unsigned long long)gptr;
  tdm_v4u g0;
  g0[0] = 1u;                                          // count=1, user D#
  g0[1] = lds_off;                                     // lds_addr
  g0[2] = (unsigned)(ga & 0xFFFFFFFFu);                // global_addr[31:0]
  g0[3] = (unsigned)((ga >> 32) & 0x1FFFFFFu)          // global_addr[56:32]
        | (2u << 30);                                  // type = 2 ("image")
  tdm_v8i g1;
  g1[0] = (int)(1u << 16);                             // data_size=1 (2B), no mcast
  g1[1] = (int)((nelem & 0xFFFFu) << 16);              // tensor_dim0[15:0]
  g1[2] = (int)(((nelem >> 16) & 0xFFFFu) | (1u << 16)); // dim0[31:16], tensor_dim1=1
  g1[3] = (int)((nelem & 0xFFFFu) << 16);              // tile_dim0
  g1[4] = 0;                                           // tile_dim1/2 unused (1-D)
  g1[5] = (int)nelem;                                  // tensor_dim0_stride
  g1[6] = 0;
  g1[7] = 0;
  tdm_v4i gz = {0, 0, 0, 0};
#if __clang_major__ >= 23
  tdm_v8i gz8 = {0, 0, 0, 0, 0, 0, 0, 0};
  __builtin_amdgcn_tensor_load_to_lds(g0, g1, gz, gz, gz8, 0);
#else
  __builtin_amdgcn_tensor_load_to_lds(g0, g1, gz, gz, 0);
#endif
}
#endif  // HAVE_TDM

// Load one 16-half WMMA fragment: per the CDNA5 16-bit A/B layout, a lane's 16
// halfs are two contiguous 8-half (16B) chunks at +0 and +16 halfs.
__device__ __forceinline__ v16h ld_frag(const _Float16* p) {
  h8_t lo = *(const h8_t*)p;
  h8_t hi = *(const h8_t*)(p + 16);
  v16h r;
#pragma unroll
  for (int i = 0; i < 8; ++i) { r[i] = lo[i]; r[i + 8] = hi[i]; }
  return r;
}

// ---------------------------------------------------------------------------
// f32 -> f16 convert (weights)
// ---------------------------------------------------------------------------
__global__ void __launch_bounds__(256)
cvt_f16_kernel(const float* __restrict__ in, _Float16* __restrict__ out, int n) {
  int i = blockIdx.x * 256 + threadIdx.x;
  if (i < n) out[i] = (_Float16)in[i];
}

// ---------------------------------------------------------------------------
// feature map transpose: [V,B,E,H,W] f32 -> [V,B,H,W,E] f16 (channels-last so
// a bilinear tap reads 512 contiguous bytes). One block per (v,b,y).
// ---------------------------------------------------------------------------
__global__ void __launch_bounds__(256)
fm_transpose_kernel(const float* __restrict__ fm, _Float16* __restrict__ fmT) {
  __shared__ __align__(16) _Float16 tile[256 * 66];   // [e][x], padded stride
  const int tid = threadIdx.x;
  const int vb  = blockIdx.x >> 6;                    // v*B + b
  const int y   = blockIdx.x & 63;
  const float* src = fm + (long)vb * (CE * CH * CW) + (long)y * CW;
#pragma unroll 4
  for (int pass = 0; pass < 64; ++pass) {
    int e = (pass << 2) + (tid >> 6);
    int x = tid & 63;
    tile[e * 66 + x] = (_Float16)src[(long)e * (CH * CW) + x];
  }
  __syncthreads();
  _Float16* dst = fmT + ((long)vb * (CH * CW) + (long)y * CW) * CE;
  for (int x = 0; x < 64; ++x)
    dst[(long)x * CE + tid] = tile[tid * 66 + x];
}

// ---------------------------------------------------------------------------
// pos-MLP (3->128->256, ReLU) then q_in = pe @ q_W.T + q_b.  One block/point.
// ---------------------------------------------------------------------------
__global__ void __launch_bounds__(128)
posq_kernel(const float* __restrict__ pts,
            const float* __restrict__ W1, const float* __restrict__ b1,
            const float* __restrict__ W2, const float* __restrict__ b2,
            const float* __restrict__ qW, const float* __restrict__ qb,
            _Float16* __restrict__ q_in) {
  __shared__ float hid[128];
  __shared__ float pe[256];
  const long m = blockIdx.x;
  const int  t = threadIdx.x;
  const float p0 = pts[m * 3 + 0], p1 = pts[m * 3 + 1], p2 = pts[m * 3 + 2];
  float hv = p0 * W1[t * 3 + 0] + p1 * W1[t * 3 + 1] + p2 * W1[t * 3 + 2] + b1[t];
  hid[t] = hv > 0.f ? hv : 0.f;
  __syncthreads();
#pragma unroll
  for (int o = 0; o < 2; ++o) {
    int e = t + o * 128;
    float acc = b2[e];
#pragma unroll 8
    for (int j = 0; j < 128; ++j) acc += hid[j] * W2[e * 128 + j];
    pe[e] = acc;
  }
  __syncthreads();
#pragma unroll
  for (int o = 0; o < 2; ++o) {
    int e = t + o * 128;
    float acc = qb[e];
#pragma unroll 8
    for (int j = 0; j < 256; ++j) acc += pe[j] * qW[e * 256 + j];
    q_in[m * CE + e] = (_Float16)acc;
  }
}

// ---------------------------------------------------------------------------
// projection + bilinear grid-sample.  grid = (M, V); 256 threads = 256 chans.
// ---------------------------------------------------------------------------
__global__ void __launch_bounds__(256)
sample_kernel(const float* __restrict__ pts, const float* __restrict__ R,
              const float* __restrict__ T,  const float* __restrict__ Km,
              const _Float16* __restrict__ fmT, _Float16* __restrict__ vf) {
  const long m = blockIdx.x;
  const int  v = blockIdx.y;
  const int  e = threadIdx.x;
  const int  b  = (int)(m >> 14);                 // N = 16384
  const int  vb = v * CB + b;
  const float* p  = pts + m * 3;
  const float* Rp = R  + (long)vb * 9;
  const float* Kp = Km + (long)vb * 9;
  const float* Tp = T  + (long)vb * 3;
  float cam[3], img[3];
#pragma unroll
  for (int j = 0; j < 3; ++j)
    cam[j] = p[0] * Rp[0 * 3 + j] + p[1] * Rp[1 * 3 + j] + p[2] * Rp[2 * 3 + j] + Tp[j];
#pragma unroll
  for (int j = 0; j < 3; ++j)
    img[j] = cam[0] * Kp[0 * 3 + j] + cam[1] * Kp[1 * 3 + j] + cam[2] * Kp[2 * 3 + j];
  const float cx = img[0] / img[2];
  const float cy = img[1] / img[2];
  const float x  = (cx + 1.f) * (CW * 0.5f) - 0.5f;
  const float y  = (cy + 1.f) * (CH * 0.5f) - 0.5f;
  const float x0f = floorf(x), y0f = floorf(y);
  const float wx1 = x - x0f, wx0 = 1.f - wx1;
  const float wy1 = y - y0f, wy0 = 1.f - wy1;
  const int x0 = (int)x0f, y0 = (int)y0f;
  const _Float16* base = fmT + (long)vb * (CH * CW * CE);
  auto tap = [&](int yy, int xx) -> float {
    if (xx < 0 || xx >= CW || yy < 0 || yy >= CH) return 0.f;
    return (float)base[((long)yy * CW + xx) * CE + e];
  };
  float acc = tap(y0,     x0)     * (wy0 * wx0)
            + tap(y0,     x0 + 1) * (wy0 * wx1)
            + tap(y0 + 1, x0)     * (wy1 * wx0)
            + tap(y0 + 1, x0 + 1) * (wy1 * wx1);
  vf[((m * CV) + v) * CE + e] = (_Float16)acc;
}

// ---------------------------------------------------------------------------
// Generic WMMA GEMM: out[r,256] = A[r,256] @ W.T + bias  (W f16 [out][in]).
// 64 rows per block (staged via TDM); 8 waves x 8 tiles; 8 WMMA per tile.
// ---------------------------------------------------------------------------
__global__ void __launch_bounds__(256)
gemm256_wmma_kernel(const _Float16* __restrict__ A, const _Float16* __restrict__ W16,
                    const float* __restrict__ bias, _Float16* __restrict__ out) {
  __shared__ __align__(16) _Float16 sA[64 * 256];
  const int  tid     = threadIdx.x;
  const long rowBase = (long)blockIdx.x * 64;
  __builtin_prefetch(W16 + (tid << 6), 0, 0);       // warm weight lines (L2)
#ifdef HAVE_TDM
  if (tid < 32) {                                   // wave 0 drives the DMA
    tdm_load_1d_f16(lds_off_of(sA), A + rowBase * CE, 64 * CE);
    __builtin_amdgcn_s_wait_tensorcnt(0);
  }
#else
  {
    const uint4* gsrc = (const uint4*)(A + rowBase * CE);
    uint4*       ldst = (uint4*)sA;
#pragma unroll
    for (int i = 0; i < 8; ++i) ldst[tid + i * 256] = gsrc[tid + i * 256];
  }
#endif
  __syncthreads();
  const int lane   = tid & 31;
  const int wave   = tid >> 5;
  const int ln     = lane & 15;
  const int koff   = (lane < 16) ? 0 : 8;   // K-half selector (A/B 16-bit layout)
  const int rowOff = (lane < 16) ? 0 : 8;   // C/D row selector
  for (int t = 0; t < 8; ++t) {
    const int tile = wave * 8 + t;          // 0..63 = 4 row-tiles x 16 col-tiles
    const int r0 = (tile >> 4) * 16;
    const int n0 = (tile & 15) * 16;
    v8f acc = {};
#pragma unroll
    for (int kk = 0; kk < 8; ++kk) {
      const int k0 = kk * 32;
      v16h a = ld_frag(&sA[(r0 + ln) * CE + k0 + koff]);
      v16h b = ld_frag(&W16[(n0 + ln) * CE + k0 + koff]);
      acc = __builtin_amdgcn_wmma_f32_16x16x32_f16(false, a, false, b,
                                                   (short)0, acc, false, false);
    }
    const float bb = bias[n0 + ln];
#pragma unroll
    for (int j = 0; j < 8; ++j)
      out[(rowBase + r0 + j + rowOff) * CE + n0 + ln] = (_Float16)(acc[j] + bb);
  }
}

// in-LDS variant used by the fused kernel (D tile written to LDS as f16)
__device__ __forceinline__ void
wmma_gemm_lds(const _Float16* sA, const _Float16* __restrict__ W16,
              const float* __restrict__ bias, _Float16* sOut, int tid) {
  const int lane   = tid & 31;
  const int wave   = tid >> 5;
  const int ln     = lane & 15;
  const int koff   = (lane < 16) ? 0 : 8;
  const int rowOff = (lane < 16) ? 0 : 8;
  for (int t = 0; t < 8; ++t) {
    const int tile = wave * 8 + t;
    const int r0 = (tile >> 4) * 16;
    const int n0 = (tile & 15) * 16;
    v8f acc = {};
#pragma unroll
    for (int kk = 0; kk < 8; ++kk) {
      const int k0 = kk * 32;
      v16h a = ld_frag(&sA[(r0 + ln) * CE + k0 + koff]);
      v16h b = ld_frag(&W16[(n0 + ln) * CE + k0 + koff]);
      acc = __builtin_amdgcn_wmma_f32_16x16x32_f16(false, a, false, b,
                                                   (short)0, acc, false, false);
    }
    const float bb = bias[n0 + ln];
#pragma unroll
    for (int j = 0; j < 8; ++j)
      sOut[(r0 + j + rowOff) * CE + n0 + ln] = (_Float16)(acc[j] + bb);
  }
}

// ---------------------------------------------------------------------------
// Fused K/V projection (WMMA, K and V tiles live only in LDS) + attention.
// Block = 8 points = 64 kv rows.  LDS: 32+32+32+4 KB = 100 KB of 320 KB.
// ---------------------------------------------------------------------------
__global__ void __launch_bounds__(256)
kv_attn_kernel(const _Float16* __restrict__ vf, const _Float16* __restrict__ qh,
               const _Float16* __restrict__ Wk16, const _Float16* __restrict__ Wv16,
               const float* __restrict__ bk, const float* __restrict__ bv,
               _Float16* __restrict__ ctx) {
  __shared__ __align__(16) _Float16 sA[64 * 256];   // vf tile
  __shared__ __align__(16) _Float16 sK[64 * 256];
  __shared__ __align__(16) _Float16 sV[64 * 256];
  __shared__ __align__(16) _Float16 sQ[8 * 256];
  const int  tid = threadIdx.x;
  const long m0  = (long)blockIdx.x * 8;            // 8 points / block
  __builtin_prefetch(Wk16 + (tid << 6), 0, 0);
  __builtin_prefetch(Wv16 + (tid << 6), 0, 0);
#ifdef HAVE_TDM
  if (tid < 32) {                                   // wave 0 drives the DMA
    tdm_load_1d_f16(lds_off_of(sA), vf + m0 * CV * CE, 64 * CE);
    tdm_load_1d_f16(lds_off_of(sQ), qh + m0 * CE, 8 * CE);
    __builtin_amdgcn_s_wait_tensorcnt(0);
  }
#else
  {
    const uint4* gA = (const uint4*)(vf + m0 * CV * CE);
    uint4*       lA = (uint4*)sA;
#pragma unroll
    for (int i = 0; i < 8; ++i) lA[tid + i * 256] = gA[tid + i * 256];
    ((uint4*)sQ)[tid] = ((const uint4*)(qh + m0 * CE))[tid];
  }
#endif
  __syncthreads();
  wmma_gemm_lds(sA, Wk16, bk, sK, tid);             // K = vf @ Wk.T + bk
  wmma_gemm_lds(sA, Wv16, bv, sV, tid);             // V = vf @ Wv.T + bv
  __syncthreads();
  // tiny-seq attention: V=8 keys, HD=32.  q identical across query views, so
  // one context row per (point, head).
  if (tid < 64) {
    const int p = tid >> 3, h = tid & 7;
    const int d0 = h * 32;
    float q[32];
#pragma unroll
    for (int d = 0; d < 32; ++d) q[d] = (float)sQ[p * CE + d0 + d];
    float s[8], mx = -1e30f;
#pragma unroll
    for (int k = 0; k < 8; ++k) {
      float acc = 0.f;
#pragma unroll
      for (int d = 0; d < 32; ++d)
        acc += q[d] * (float)sK[(p * 8 + k) * CE + d0 + d];
      s[k] = acc * 0.17677669529663687f;            // 1/sqrt(32)
      mx = s[k] > mx ? s[k] : mx;
    }
    float sum = 0.f;
#pragma unroll
    for (int k = 0; k < 8; ++k) { s[k] = __expf(s[k] - mx); sum += s[k]; }
    const float inv = 1.f / sum;
#pragma unroll
    for (int d = 0; d < 32; ++d) {
      float c = 0.f;
#pragma unroll
      for (int k = 0; k < 8; ++k)
        c += s[k] * (float)sV[(p * 8 + k) * CE + d0 + d];
      ctx[(m0 + p) * CE + d0 + d] = (_Float16)(c * inv);
    }
  }
}

// ---------------------------------------------------------------------------
// LayerNorm with residual:  out = LN(a + r) * g + b   (f16 in/out)
// ---------------------------------------------------------------------------
__global__ void __launch_bounds__(256)
ln_res_kernel(const _Float16* __restrict__ a, const _Float16* __restrict__ r,
              const float* __restrict__ g, const float* __restrict__ bta,
              _Float16* __restrict__ out) {
  __shared__ float red[256];
  const long m = blockIdx.x;
  const int  t = threadIdx.x;
  const float x = (float)a[m * CE + t] + (float)r[m * CE + t];
  red[t] = x; __syncthreads();
  for (int s = 128; s > 0; s >>= 1) { if (t < s) red[t] += red[t + s]; __syncthreads(); }
  const float mean = red[0] * (1.f / 256.f); __syncthreads();
  red[t] = (x - mean) * (x - mean); __syncthreads();
  for (int s = 128; s > 0; s >>= 1) { if (t < s) red[t] += red[t + s]; __syncthreads(); }
  const float var = red[0] * (1.f / 256.f);
  out[m * CE + t] = (_Float16)((x - mean) * rsqrtf(var + 1e-5f) * g[t] + bta[t]);
}

// ---------------------------------------------------------------------------
// Final LN + outputs.  aggregated == LN2 row (final identical across views,
// view_attention == 1/V exactly).
// ---------------------------------------------------------------------------
__global__ void __launch_bounds__(256)
final_kernel(const _Float16* __restrict__ y, const float* __restrict__ g,
             const float* __restrict__ bta, float* __restrict__ agg,
             float* __restrict__ vatt) {
  __shared__ float red[256];
  const long m = blockIdx.x;
  const int  t = threadIdx.x;
  const float x = (float)y[m * CE + t];
  red[t] = x; __syncthreads();
  for (int s = 128; s > 0; s >>= 1) { if (t < s) red[t] += red[t + s]; __syncthreads(); }
  const float mean = red[0] * (1.f / 256.f); __syncthreads();
  red[t] = (x - mean) * (x - mean); __syncthreads();
  for (int s = 128; s > 0; s >>= 1) { if (t < s) red[t] += red[t + s]; __syncthreads(); }
  const float var = red[0] * (1.f / 256.f);
  agg[m * CE + t] = (x - mean) * rsqrtf(var + 1e-5f) * g[t] + bta[t];
  if (t < CV) vatt[m * CV + t] = 0.125f;
}

// ---------------------------------------------------------------------------
// host launcher
// ---------------------------------------------------------------------------
extern "C" void kernel_launch(void* const* d_in, const int* in_sizes, int n_in,
                              void* d_out, int out_size, void* d_ws, size_t ws_size,
                              hipStream_t stream) {
  const float* pts   = (const float*)d_in[0];
  const float* fm    = (const float*)d_in[1];
  const float* Rm    = (const float*)d_in[2];
  const float* Tm    = (const float*)d_in[3];
  const float* Km    = (const float*)d_in[4];
  const float* pW1   = (const float*)d_in[5];
  const float* pb1   = (const float*)d_in[6];
  const float* pW2   = (const float*)d_in[7];
  const float* pb2   = (const float*)d_in[8];
  const float* qW    = (const float*)d_in[9];
  const float* qb    = (const float*)d_in[10];
  const float* Wq    = (const float*)d_in[11];
  const float* bq    = (const float*)d_in[12];
  const float* Wk    = (const float*)d_in[13];
  const float* bk    = (const float*)d_in[14];
  const float* Wv    = (const float*)d_in[15];
  const float* bv    = (const float*)d_in[16];
  const float* Wo    = (const float*)d_in[17];
  const float* bo    = (const float*)d_in[18];
  const float* outW  = (const float*)d_in[19];
  const float* outb  = (const float*)d_in[20];
  const float* n1g   = (const float*)d_in[21];
  const float* n1b   = (const float*)d_in[22];
  const float* n2g   = (const float*)d_in[23];
  const float* n2b   = (const float*)d_in[24];
  (void)in_sizes; (void)n_in; (void)out_size; (void)ws_size;

  // workspace layout (bytes, 256-aligned); peak ~219 MB
  char* ws = (char*)d_ws;
  size_t off = 0;
  auto carve = [&](size_t bytes) { char* p = ws + off; off = (off + bytes + 255) & ~(size_t)255; return p; };
  _Float16* Wq16  = (_Float16*)carve(CE * CE * 2);
  _Float16* Wk16  = (_Float16*)carve(CE * CE * 2);
  _Float16* Wv16  = (_Float16*)carve(CE * CE * 2);
  _Float16* Wo16  = (_Float16*)carve(CE * CE * 2);
  _Float16* oW16  = (_Float16*)carve(CE * CE * 2);
  _Float16* fmT   = (_Float16*)carve((size_t)CV * CB * CH * CW * CE * 2);   // 33.5 MB
  _Float16* q_in  = (_Float16*)carve((size_t)CM * CE * 2);                  // 16.8 MB
  _Float16* qh    = (_Float16*)carve((size_t)CM * CE * 2);
  _Float16* ctx16 = (_Float16*)carve((size_t)CM * CE * 2);
  _Float16* vfbuf = (_Float16*)carve((size_t)CMV * CE * 2);                 // 134 MB
  // vf region is dead after kv_attn; reuse it for the tail activations
  _Float16* att0  = vfbuf;
  _Float16* x1    = vfbuf + (size_t)CM * CE;
  _Float16* y16   = vfbuf + (size_t)2 * CM * CE;

  float* aggr = (float*)d_out;                         // [B,N,E]
  float* vatt = (float*)d_out + (size_t)CM * CE;       // [B,N,V]

  const int nW = CE * CE;
  cvt_f16_kernel<<<(nW + 255) / 256, 256, 0, stream>>>(Wq,   Wq16, nW);
  cvt_f16_kernel<<<(nW + 255) / 256, 256, 0, stream>>>(Wk,   Wk16, nW);
  cvt_f16_kernel<<<(nW + 255) / 256, 256, 0, stream>>>(Wv,   Wv16, nW);
  cvt_f16_kernel<<<(nW + 255) / 256, 256, 0, stream>>>(Wo,   Wo16, nW);
  cvt_f16_kernel<<<(nW + 255) / 256, 256, 0, stream>>>(outW, oW16, nW);

  fm_transpose_kernel<<<CV * CB * CH, 256, 0, stream>>>(fm, fmT);

  posq_kernel<<<(int)CM, 128, 0, stream>>>(pts, pW1, pb1, pW2, pb2, qW, qb, q_in);

  // qh = q_in @ Wq.T + bq   (one row per point; identical across views)
  gemm256_wmma_kernel<<<(int)(CM / 64), 256, 0, stream>>>(q_in, Wq16, bq, qh);

  sample_kernel<<<dim3((unsigned)CM, CV), 256, 0, stream>>>(pts, Rm, Tm, Km, fmT, vfbuf);

  // fused K/V projection + attention -> ctx [M,E]
  kv_attn_kernel<<<(int)(CM / 8), 256, 0, stream>>>(vfbuf, qh, Wk16, Wv16, bk, bv, ctx16);

  // attended = ctx @ Wo.T + bo
  gemm256_wmma_kernel<<<(int)(CM / 64), 256, 0, stream>>>(ctx16, Wo16, bo, att0);
  // x1 = LN(attended + q_in)
  ln_res_kernel<<<(int)CM, 256, 0, stream>>>(att0, q_in, n1g, n1b, x1);
  // y = x1 @ out_W.T + out_b
  gemm256_wmma_kernel<<<(int)(CM / 64), 256, 0, stream>>>(x1, oW16, outb, y16);
  // aggregated = LN2(y); view_attention = 1/V
  final_kernel<<<(int)CM, 256, 0, stream>>>(y16, n2g, n2b, aggr, vatt);
}
